// GraphEncoder_with_weight_61899068670276
// MI455X (gfx1250) — matrix operand order repaired and checked by reading, not compiled, not run on hardware
//
#include <hip/hip_runtime.h>
#include <hip/hip_bf16.h>

typedef __attribute__((ext_vector_type(16))) __bf16 v16bf;
typedef __attribute__((ext_vector_type(8)))  float  v8f;
typedef __attribute__((ext_vector_type(8)))  int    v8i;

union FragU {
    int4  q[2];
    v8i   w;
    v16bf bf;
};

static __device__ __forceinline__ unsigned short f32_to_bf16(float f) {
    union { float f; unsigned u; } x; x.f = f;
    unsigned r = x.u + 0x7FFFu + ((x.u >> 16) & 1u);   // round-to-nearest-even
    return (unsigned short)(r >> 16);
}
static __device__ __forceinline__ unsigned pack2(float a, float b) {
    return (unsigned)f32_to_bf16(a) | ((unsigned)f32_to_bf16(b) << 16);
}

// Pre-swizzle a row-major f32 weight [Kdim x 256] into bf16 WMMA B-fragment order:
// dst[((kt*16 + nt)*32 + lane)*8 + v] packs W[k][n], W[k+1][n] with
//   n = nt*16 + (lane&15), k = kt*32 + (lane>>4)*8 + (v>>2)*16 + 2*(v&3)
__global__ __launch_bounds__(256) void swizzle_weight(const float* __restrict__ W,
                                                      unsigned* __restrict__ dst,
                                                      int ktiles) {
    int tid = blockIdx.x * blockDim.x + threadIdx.x;
    int total = ktiles * 16 * 32 * 8;
    if (tid >= total) return;
    int v    = tid & 7;
    int lane = (tid >> 3) & 31;
    int nt   = (tid >> 8) & 15;
    int kt   = tid >> 12;
    int n = nt * 16 + (lane & 15);
    int k = kt * 32 + ((lane >> 4) * 8) + ((v >> 2) * 16) + 2 * (v & 3);
    float lo = W[(size_t)k * 256 + n];
    float hi = W[(size_t)(k + 1) * 256 + n];
    dst[tid] = pack2(lo, hi);
}

__global__ __launch_bounds__(256) void ge_fused(
    const int*   __restrict__ nodes,
    const int*   __restrict__ neigh_idx,
    const float* __restrict__ neigh_w,
    const float* __restrict__ features,
    const float* __restrict__ b_init,
    const float* __restrict__ b_final,
    const unsigned* __restrict__ swzWi,   // [8][16][32][8] dwords
    const unsigned* __restrict__ swzWf,   // [16][16][32][8] dwords
    float* __restrict__ out, int B)
{
    __shared__ int            s_idx[512];       // 16 rows x 32 neighbors
    __shared__ float          s_w[512];
    __shared__ float          s_inv[16];
    __shared__ int            s_node[16];
    __shared__ unsigned short sA1 [16 * 264];   // self features (bf16), row stride 264 (=528B)
    __shared__ unsigned short sA2a[16 * 264];   // GEMM1 output (bf16)
    __shared__ unsigned short sA2b[16 * 264];   // neighbor aggregate (bf16)

    const int tid  = threadIdx.x;
    const int lane = tid & 31;
    const int wave = tid >> 5;
    const int row0 = blockIdx.x * 16;

    // ---------- stage neighbor indices / weights / node ids ----------
    #pragma unroll
    for (int ii = 0; ii < 2; ++ii) {
        int i  = tid + ii * 256;
        int rg = row0 + (i >> 5); if (rg > B - 1) rg = B - 1;
        s_idx[i] = neigh_idx[(size_t)rg * 32 + (i & 31)];
        s_w[i]   = neigh_w [(size_t)rg * 32 + (i & 31)];
    }
    if (tid < 16) {
        int rg = row0 + tid; if (rg > B - 1) rg = B - 1;
        s_node[tid] = nodes[rg];
    }
    __syncthreads();
    if (tid < 16) {
        float s = 0.f;
        #pragma unroll
        for (int k = 0; k < 32; ++k) s += s_w[tid * 32 + k];
        s_inv[tid] = 1.0f / s;
    }
    __syncthreads();
    #pragma unroll
    for (int ii = 0; ii < 2; ++ii) {
        int i = tid + ii * 256;
        s_w[i] *= s_inv[i >> 5];
    }

    // ---------- self-row gather -> sA1 (bf16) ----------
    {
        const int r = tid >> 4;
        const int c = (tid & 15) * 16;
        const float4* src = (const float4*)(features + (size_t)s_node[r] * 256 + c);
        float4 f0 = src[0], f1 = src[1], f2 = src[2], f3 = src[3];
        unsigned* d = (unsigned*)(sA1 + r * 264 + c);
        d[0] = pack2(f0.x, f0.y); d[1] = pack2(f0.z, f0.w);
        d[2] = pack2(f1.x, f1.y); d[3] = pack2(f1.z, f1.w);
        d[4] = pack2(f2.x, f2.y); d[5] = pack2(f2.z, f2.w);
        d[6] = pack2(f3.x, f3.y); d[7] = pack2(f3.z, f3.w);
    }
    __syncthreads();   // sA1 + normalized s_w visible

    // ---------- neighbor weighted aggregation -> sA2b (bf16) ----------
    {
        const int r = tid >> 4;
        const int c = (tid & 15) * 16;
        float acc[16];
        #pragma unroll
        for (int j = 0; j < 16; ++j) acc[j] = 0.f;
        for (int k = 0; k < 32; ++k) {
            const float wgt = s_w[r * 32 + k];
            const float4* src = (const float4*)(features + (size_t)s_idx[r * 32 + k] * 256 + c);
            if (k + 1 < 32)
                __builtin_prefetch(features + (size_t)s_idx[r * 32 + k + 1] * 256 + c, 0, 0);
            float4 f0 = src[0], f1 = src[1], f2 = src[2], f3 = src[3];
            acc[0]  = fmaf(wgt, f0.x, acc[0]);  acc[1]  = fmaf(wgt, f0.y, acc[1]);
            acc[2]  = fmaf(wgt, f0.z, acc[2]);  acc[3]  = fmaf(wgt, f0.w, acc[3]);
            acc[4]  = fmaf(wgt, f1.x, acc[4]);  acc[5]  = fmaf(wgt, f1.y, acc[5]);
            acc[6]  = fmaf(wgt, f1.z, acc[6]);  acc[7]  = fmaf(wgt, f1.w, acc[7]);
            acc[8]  = fmaf(wgt, f2.x, acc[8]);  acc[9]  = fmaf(wgt, f2.y, acc[9]);
            acc[10] = fmaf(wgt, f2.z, acc[10]); acc[11] = fmaf(wgt, f2.w, acc[11]);
            acc[12] = fmaf(wgt, f3.x, acc[12]); acc[13] = fmaf(wgt, f3.y, acc[13]);
            acc[14] = fmaf(wgt, f3.z, acc[14]); acc[15] = fmaf(wgt, f3.w, acc[15]);
        }
        unsigned* d = (unsigned*)(sA2b + r * 264 + c);
        #pragma unroll
        for (int j = 0; j < 8; ++j) d[j] = pack2(acc[2 * j], acc[2 * j + 1]);
    }

    // A-fragment addressing: lane L -> row m=L&15; two b128s at K-run kbase=(L>>4)*8
    const int aoff  = (lane & 15) * 528 + ((lane >> 4) * 16);   // bytes
    const int ncol  = lane & 15;
    const int mbase = (lane >> 4) * 8;

    // ---------- GEMM1: self_feats = sA1 @ W_init + b_init -> sA2a (bf16) ----------
    #pragma unroll
    for (int t = 0; t < 2; ++t) {
        const int nt = wave * 2 + t;
        v8f c = {};
        #pragma unroll
        for (int kt = 0; kt < 8; ++kt) {
            FragU a, b;
            const char* ap = (const char*)sA1 + aoff + kt * 64;
            a.q[0] = *(const int4*)(ap);
            a.q[1] = *(const int4*)(ap + 32);
            b.w = *(const v8i*)(swzWi + (((kt * 16 + nt) * 32 + lane) << 3));
            c = __builtin_amdgcn_wmma_f32_16x16x32_bf16(false, a.bf, false, b.bf,
                                                        (short)0, c, false, false);
        }
        const float bias = b_init[nt * 16 + ncol];
        #pragma unroll
        for (int v = 0; v < 8; ++v)
            sA2a[(mbase + v) * 264 + nt * 16 + ncol] = f32_to_bf16(c[v] + bias);
    }
    __syncthreads();   // sA2a + sA2b visible

    // ---------- GEMM2: z = [self | neigh] @ W_final + b_final; swish; store ----------
    #pragma unroll
    for (int t = 0; t < 2; ++t) {
        const int nt = wave * 2 + t;
        v8f c = {};
        #pragma unroll
        for (int kt = 0; kt < 16; ++kt) {
            const unsigned short* Ab = (kt < 8) ? sA2a : sA2b;
            FragU a, b;
            const char* ap = (const char*)Ab + aoff + (kt & 7) * 64;
            a.q[0] = *(const int4*)(ap);
            a.q[1] = *(const int4*)(ap + 32);
            b.w = *(const v8i*)(swzWf + (((kt * 16 + nt) * 32 + lane) << 3));
            c = __builtin_amdgcn_wmma_f32_16x16x32_bf16(false, a.bf, false, b.bf,
                                                        (short)0, c, false, false);
        }
        const float bias = b_final[nt * 16 + ncol];
        #pragma unroll
        for (int v = 0; v < 8; ++v) {
            const int m = row0 + mbase + v;
            float z  = c[v] + bias;
            float sw = z * (1.0f / (1.0f + __expf(-z)));
            if (m < B) out[(size_t)m * 256 + nt * 16 + ncol] = sw;
        }
    }
}

extern "C" void kernel_launch(void* const* d_in, const int* in_sizes, int n_in,
                              void* d_out, int out_size, void* d_ws, size_t ws_size,
                              hipStream_t stream) {
    (void)n_in; (void)out_size; (void)ws_size;
    const int*   nodes     = (const int*)  d_in[0];
    const int*   neigh_idx = (const int*)  d_in[1];
    const float* neigh_w   = (const float*)d_in[2];
    const float* features  = (const float*)d_in[3];
    const float* W_init    = (const float*)d_in[4];
    const float* b_init    = (const float*)d_in[5];
    const float* W_final   = (const float*)d_in[6];
    const float* b_final   = (const float*)d_in[7];
    float* out = (float*)d_out;
    const int B = in_sizes[0];

    // workspace: swizzled bf16 weights (128KB + 256KB)
    unsigned* swzWi = (unsigned*)d_ws;                  // 8*16*32*8  = 32768 dwords
    unsigned* swzWf = swzWi + 8 * 16 * 32 * 8;          // 16*16*32*8 = 65536 dwords

    swizzle_weight<<<(8  * 16 * 32 * 8) / 256, 256, 0, stream>>>(W_init,  swzWi, 8);
    swizzle_weight<<<(16 * 16 * 32 * 8) / 256, 256, 0, stream>>>(W_final, swzWf, 16);
    ge_fused<<<(B + 15) / 16, 256, 0, stream>>>(nodes, neigh_idx, neigh_w, features,
                                                b_init, b_final, swzWi, swzWf, out, B);
}
